// TransformerVQVAE_12979391169319
// MI455X (gfx1250) — matrix operand order repaired and checked by reading, not compiled, not run on hardware
//
#include <hip/hip_runtime.h>
#include <hip/hip_bf16.h>
#include <math.h>

// ---------------------------------------------------------------------------
// Transformer VQ-VAE forward on gfx1250:
//   * f16 activations + f16 weights (pre-converted), f32 WMMA accumulation
//   * v_wmma_f32_16x16x32_f16 for all GEMMs + attention
//   * Tensor Data Mover (tensor_load_to_lds) stages all LDS tiles, with
//     hardware zero-fill for K tails / padded rows and hardware LDS padding
//     for bank-conflict-free strides. s_wait_tensorcnt + barrier for sync.
// ---------------------------------------------------------------------------

typedef __attribute__((ext_vector_type(16))) _Float16 v16h;
typedef __attribute__((ext_vector_type(8)))  _Float16 v8h;
typedef __attribute__((ext_vector_type(8)))  float    v8f;

typedef unsigned int       u32;
typedef unsigned long long u64;
typedef __attribute__((ext_vector_type(4))) u32 v4u;
typedef __attribute__((ext_vector_type(4))) int v4i;
typedef __attribute__((ext_vector_type(8))) int v8i_t;

#define B_     2048
#define S_     64
#define C_     142
#define D_     256
#define H_     8
#define DH_    32
#define FF_    512
#define L_     3
#define E_     128
#define KCODE_ 1024
#define T_     65   // S + 1 (cls token)

// --------------------------- TDM availability ------------------------------
#define HAVE_TDM 0
#if defined(__HIP_DEVICE_COMPILE__)
#  if __has_builtin(__builtin_amdgcn_tensor_load_to_lds)
#    undef HAVE_TDM
#    define HAVE_TDM 1
#  endif
#endif

#if HAVE_TDM
// 2D f16 tile load: tile_d0 x tile_d1 elements, row stride stride0 elements.
// OOB (x >= tensor_d0 or y >= tensor_d1) reads return zero (hardware fill).
// pad_i/pad_a: LDS padding codes (interval = 2^(pad_i+1) DW, amount = pad_a+1 DW).
__device__ __forceinline__ void tdm_load_2d(u32 lds_byte, const _Float16* gptr,
                                            u32 tile_d0, u32 tile_d1,
                                            u32 tensor_d0, u32 tensor_d1,
                                            u64 stride0,
                                            u32 pad_i, u32 pad_a, u32 pad_en)
{
    u64 ga = (u64)(size_t)gptr;
    v4u g0;
    g0.x = 1u;                                            // count=1 (user mode)
    g0.y = lds_byte;                                      // lds_addr [63:32]
    g0.z = (u32)(ga & 0xFFFFFFFFull);                     // global_addr [95:64]
    g0.w = (u32)((ga >> 32) & 0x1FFFFFFull) | (2u << 30); // addr hi + type=2

    u32 w0 = (1u << 16) | (pad_en << 20) | (pad_i << 22) | (pad_a << 25); // 2B
    u32 w1 = (tensor_d0 & 0xFFFFu) << 16;                 // tensor_dim0[15:0]
    u32 w2 = (tensor_d0 >> 16) | ((tensor_d1 & 0xFFFFu) << 16);
    u32 w3 = (tensor_d1 >> 16) | (tile_d0 << 16);         // + tile_dim0
    u32 w4 = (tile_d1 & 0xFFFFu);                         // tile_dim1, tile_dim2=0
    u32 w5 = (u32)(stride0 & 0xFFFFFFFFull);              // dim0_stride[31:0]
    u32 w6 = (u32)((stride0 >> 32) & 0xFFFFull);          // dim0_stride[47:32]
    u32 w7 = 0;
    v8i_t g1 = { (int)w0, (int)w1, (int)w2, (int)w3,
                 (int)w4, (int)w5, (int)w6, (int)w7 };
    v4i z4 = {0, 0, 0, 0};
#if defined(__clang_major__) && (__clang_major__ >= 23)
    v8i_t z8 = {0, 0, 0, 0, 0, 0, 0, 0};
    __builtin_amdgcn_tensor_load_to_lds(g0, g1, z4, z4, z8, 0);
#else
    __builtin_amdgcn_tensor_load_to_lds(g0, g1, z4, z4, 0);
#endif
}
#endif // HAVE_TDM

__device__ __forceinline__ float gelu_exact(float v) {
    return 0.5f * v * (1.0f + erff(v * 0.70710678118654752f));
}

// ---------------------------------------------------------------------------
// WMMA GEMM: C[M,N] = A[M,K] @ W[N,K]^T + bias[N]. A,W are f16 row-major.
// REQUIRES M%64==0 and N%64==0 (true for every call site). K tail handled by
// TDM zero-fill (or fallback path).
// Epilogues: 0 bias | 1 bias+GELU | 2 bias+PE(f32 aux)+cls row remap
//            3 bias+residual(f16 aux). out_half selects f16/f32 output.
// Block: 256 threads = 8 waves, tile 64x64, K step 32.
// ---------------------------------------------------------------------------
__global__ __launch_bounds__(256)
void gemm_wmma_kernel(const _Float16* __restrict__ A, int lda,
                      const _Float16* __restrict__ W, int ldw,
                      const float* __restrict__ bias,
                      const void* __restrict__ aux,
                      void* __restrict__ Cout, int ldc,
                      int M, int N, int Kd, int epilogue, int out_half)
{
    __shared__ _Float16 As[64][40];   // 32 data halves + 8 pad halves per row
    __shared__ _Float16 Ws[64][40];

    const int m0   = blockIdx.x * 64;
    const int n0   = blockIdx.y * 64;
    const int tid  = threadIdx.x;
    const int wave = tid >> 5;
    const int lane = tid & 31;
    const int half = lane >> 4;
    const int ln16 = lane & 15;
    const int wm   = (wave & 3) * 16;
    const int wn   = (wave >> 2) * 32;

    v8f c0 = {}; v8f c1 = {};

    for (int k0 = 0; k0 < Kd; k0 += 32) {
        __syncthreads();   // previous tile fully consumed
#if HAVE_TDM
        if (tid == 0) {
            // interval 16 DW (code 3), amount 4 DW (code 3) -> row stride 40 halves
            tdm_load_2d((u32)(size_t)&As[0][0], A + (size_t)m0 * lda + k0,
                        32, 64, (u32)(Kd - k0), 64, (u64)lda, 3, 3, 1);
            tdm_load_2d((u32)(size_t)&Ws[0][0], W + (size_t)n0 * ldw + k0,
                        32, 64, (u32)(Kd - k0), 64, (u64)ldw, 3, 3, 1);
        }
        if (tid < 32) __builtin_amdgcn_s_wait_tensorcnt(0);
#else
        int kleft = Kd - k0;
        if (kleft >= 32) {
            for (int i = tid; i < 64 * 4; i += 256) {       // 4 x v8h per row
                int r = i >> 2, c8 = (i & 3) * 8;
                *(v8h*)&As[r][c8] = *(const v8h*)&A[(size_t)(m0 + r) * lda + k0 + c8];
                *(v8h*)&Ws[r][c8] = *(const v8h*)&W[(size_t)(n0 + r) * ldw + k0 + c8];
            }
        } else {
            for (int i = tid; i < 64 * 32; i += 256) {
                int r = i >> 5, cc = i & 31;
                As[r][cc] = (cc < kleft) ? A[(size_t)(m0 + r) * lda + k0 + cc]
                                         : (_Float16)0.f;
                Ws[r][cc] = (cc < kleft) ? W[(size_t)(n0 + r) * ldw + k0 + cc]
                                         : (_Float16)0.f;
            }
        }
#endif
        __syncthreads();

        // Fragments use k = half*16 + i in A and B (each run of 16 halves is contiguous ->
        // compiler emits ds_load_b128 pairs).
        v16h a, b0, b1;
#pragma unroll
        for (int i = 0; i < 16; ++i) {
            int ka = half * 16 + i;
            a[i]  = As[wm + ln16][ka];
            int kb = half * 16 + i;
            b0[i] = Ws[wn + ln16][kb];
            b1[i] = Ws[wn + 16 + ln16][kb];
        }
        c0 = __builtin_amdgcn_wmma_f32_16x16x32_f16(false, a, false, b0,
                                                    (short)0, c0, false, false);
        c1 = __builtin_amdgcn_wmma_f32_16x16x32_f16(false, a, false, b1,
                                                    (short)0, c1, false, false);
    }

    // Epilogue (M,N multiples of 64: no bounds checks, EXEC stays full).
#pragma unroll
    for (int r = 0; r < 8; ++r) {
        int gm = m0 + wm + r + half * 8;
#pragma unroll
        for (int sub = 0; sub < 2; ++sub) {
            int gn = n0 + wn + sub * 16 + ln16;
            float v = (sub ? c1[r] : c0[r]) + bias[gn];
            size_t orow = (size_t)gm;
            if (epilogue == 2) {
                v += ((const float*)aux)[(size_t)(gm % S_) * ldc + gn];
                orow = (size_t)(gm / S_) * T_ + (gm % S_) + 1;
            } else if (epilogue == 3) {
                v += (float)((const _Float16*)aux)[(size_t)gm * ldc + gn];
            } else if (epilogue == 1) {
                v = gelu_exact(v);
            }
            if (out_half) ((_Float16*)Cout)[orow * ldc + gn] = (_Float16)v;
            else          ((float*)Cout)[orow * ldc + gn]    = v;
        }
    }
}

// ---------------------------------------------------------------------------
// Fused attention, one block per (batch, head). 160 threads = 5 waves.
// T=65 padded to 80 score rows (TDM zero-fills rows >= 65); P@V pads K to 96.
// qkv layout (f16): [B, T, 3*D]; o layout (f16): [B, T, D].
// ---------------------------------------------------------------------------
__global__ __launch_bounds__(160)
void attn_kernel(const _Float16* __restrict__ qkv, _Float16* __restrict__ o)
{
    __shared__ _Float16 Qs[80][36];   // 32 data + 4 pad halves per row
    __shared__ _Float16 Ksh[80][36];
    __shared__ _Float16 Vs[96][36];
    __shared__ _Float16 Ps[5][16][104];

    const int bh   = blockIdx.x;
    const int b    = bh / H_;
    const int hh   = bh % H_;
    const int tid  = threadIdx.x;
    const int wave = tid >> 5;
    const int lane = tid & 31;
    const int half = lane >> 4;
    const int ln16 = lane & 15;

    const size_t baseq = (size_t)b * T_ * (3 * D_) + (size_t)hh * DH_;

#if HAVE_TDM
    if (tid == 0) {
        // interval 16 DW (code 3), amount 2 DW (code 1) -> row stride 36 halves
        tdm_load_2d((u32)(size_t)&Qs[0][0],  qkv + baseq,          32, 80, 32, T_, 3 * D_, 3, 1, 1);
        tdm_load_2d((u32)(size_t)&Ksh[0][0], qkv + baseq + D_,     32, 80, 32, T_, 3 * D_, 3, 1, 1);
        tdm_load_2d((u32)(size_t)&Vs[0][0],  qkv + baseq + 2 * D_, 32, 96, 32, T_, 3 * D_, 3, 1, 1);
    }
    if (tid < 32) __builtin_amdgcn_s_wait_tensorcnt(0);
#else
    for (int i = tid; i < 96 * 4; i += 160) {       // 4 x v8h per row
        int t = i >> 2, c8 = (i & 3) * 8;
        v8h z = {};
        if (t < T_) {
            size_t idx = baseq + (size_t)t * (3 * D_) + c8;
            *(v8h*)&Qs[t][c8]  = *(const v8h*)&qkv[idx];
            *(v8h*)&Ksh[t][c8] = *(const v8h*)&qkv[idx + D_];
            *(v8h*)&Vs[t][c8]  = *(const v8h*)&qkv[idx + 2 * D_];
        } else {
            if (t < 80) { *(v8h*)&Qs[t][c8] = z; *(v8h*)&Ksh[t][c8] = z; }
            *(v8h*)&Vs[t][c8] = z;
        }
    }
#endif
    __syncthreads();

    // ---- scores = Q @ K^T / sqrt(DH); wave owns 16-row tile ----
    const int m0 = wave * 16;
    v16h a;
#pragma unroll
    for (int i = 0; i < 16; ++i) {
        int ka = half * 16 + i;
        a[i] = Qs[m0 + ln16][ka];
    }
    v8f sc[5];
#pragma unroll
    for (int t = 0; t < 5; ++t) {
        v16h bq;
#pragma unroll
        for (int i = 0; i < 16; ++i)
            bq[i] = Ksh[t * 16 + ln16][half * 16 + i];
        v8f z = {};
        sc[t] = __builtin_amdgcn_wmma_f32_16x16x32_f16(false, a, false, bq,
                                                       (short)0, z, false, false);
    }

    // ---- softmax over 80 cols (mask >= 65) ----
    const float scale = 0.17677669529663687f;   // 1/sqrt(32)
#pragma unroll
    for (int r = 0; r < 8; ++r) {
        float vals[5];
        float mx = -1e30f;
#pragma unroll
        for (int t = 0; t < 5; ++t) {
            int n = t * 16 + ln16;
            float v = sc[t][r] * scale;
            if (n >= T_) v = -1e30f;
            vals[t] = v;
            mx = fmaxf(mx, v);
        }
        for (int off = 1; off < 16; off <<= 1) mx = fmaxf(mx, __shfl_xor(mx, off, 32));
        float sum = 0.f;
#pragma unroll
        for (int t = 0; t < 5; ++t) {
            float e = __expf(vals[t] - mx);
            if (t * 16 + ln16 >= T_) e = 0.f;
            vals[t] = e;
            sum += e;
        }
        for (int off = 1; off < 16; off <<= 1) sum += __shfl_xor(sum, off, 32);
        float inv = 1.0f / sum;
        int mrow = r + half * 8;
#pragma unroll
        for (int t = 0; t < 5; ++t)
            Ps[wave][mrow][t * 16 + ln16] = (_Float16)(vals[t] * inv);
    }
    // zero-pad P columns 80..95 (wave-private; DS ops are in-order per wave)
    for (int i = lane; i < 16 * 16; i += 32)
        Ps[wave][i >> 4][80 + (i & 15)] = (_Float16)0.f;
    __syncthreads();

    // ---- out = P @ V (16 x 32 per wave), K = 96 ----
    v8f o0 = {}, o1 = {};
#pragma unroll
    for (int ks = 0; ks < 96; ks += 32) {
        v16h pa, bv0, bv1;
#pragma unroll
        for (int i = 0; i < 16; ++i) {
            int ka = ks + half * 16 + i;
            pa[i]  = Ps[wave][ln16][ka];
            int kb = ks + half * 16 + i;
            bv0[i] = Vs[kb][ln16];
            bv1[i] = Vs[kb][16 + ln16];
        }
        o0 = __builtin_amdgcn_wmma_f32_16x16x32_f16(false, pa, false, bv0,
                                                    (short)0, o0, false, false);
        o1 = __builtin_amdgcn_wmma_f32_16x16x32_f16(false, pa, false, bv1,
                                                    (short)0, o1, false, false);
    }

#pragma unroll
    for (int r = 0; r < 8; ++r) {
        int t = m0 + r + half * 8;
        if (t >= T_) continue;
        size_t ob = ((size_t)b * T_ + t) * D_ + (size_t)hh * DH_;
        o[ob + ln16]      = (_Float16)o0[r];
        o[ob + 16 + ln16] = (_Float16)o1[r];
    }
}

// ---------------------------------------------------------------------------
// LayerNorm over rows (one wave per row), f16 in/out, optional fused GELU.
// ---------------------------------------------------------------------------
__global__ __launch_bounds__(256)
void ln_kernel(const _Float16* __restrict__ X, size_t in_stride,
               const float* __restrict__ g, const float* __restrict__ bvec,
               _Float16* __restrict__ Y, size_t out_stride,
               int rows, int cols, int gelu)
{
    int row  = blockIdx.x * 8 + (threadIdx.x >> 5);
    int lane = threadIdx.x & 31;
    if (row >= rows) return;
    const _Float16* x = X + (size_t)row * in_stride;
    float s = 0.f, s2 = 0.f;
    for (int c = lane; c < cols; c += 32) { float v = (float)x[c]; s += v; s2 += v * v; }
    for (int off = 1; off < 32; off <<= 1) {
        s  += __shfl_xor(s, off, 32);
        s2 += __shfl_xor(s2, off, 32);
    }
    float mean = s / cols;
    float var  = s2 / cols - mean * mean;
    float rstd = rsqrtf(var + 1e-5f);
    _Float16* y = Y + (size_t)row * out_stride;
    for (int c = lane; c < cols; c += 32) {
        float v = ((float)x[c] - mean) * rstd * g[c] + bvec[c];
        if (gelu) v = gelu_exact(v);
        y[c] = (_Float16)v;
    }
}

// fp32 -> f16 conversion (inputs / weights)
__global__ void cvt_kernel(const float* __restrict__ src, _Float16* __restrict__ dst,
                           size_t n)
{
    size_t i = (size_t)blockIdx.x * blockDim.x + threadIdx.x;
    if (i < n) dst[i] = (_Float16)src[i];
}

// Sinusoidal positional encoding [S, D] (f32; added in GEMM epilogue 2).
__global__ void pe_kernel(float* __restrict__ pe)
{
    int i = blockIdx.x * blockDim.x + threadIdx.x;
    if (i >= S_ * (D_ / 2)) return;
    int s = i / (D_ / 2), j = i % (D_ / 2);
    float div = __expf((2.0f * j) * (-9.210340371976184f / (float)D_));
    float ang = (float)s * div;
    pe[s * D_ + 2 * j]     = __sinf(ang);
    pe[s * D_ + 2 * j + 1] = __cosf(ang);
}

__global__ void cls_kernel(const float* __restrict__ cls, _Float16* __restrict__ h)
{
    int i = blockIdx.x * blockDim.x + threadIdx.x;
    if (i >= B_ * D_) return;
    int b = i / D_, d = i % D_;
    h[(size_t)b * T_ * D_ + d] = (_Float16)cls[d];
}

__global__ void init_kernel(float* __restrict__ hist, float* __restrict__ accum)
{
    int i = blockIdx.x * blockDim.x + threadIdx.x;
    if (i < KCODE_) hist[i] = 0.f;
    if (i == 0) accum[0] = 0.f;
}

// ---------------------------------------------------------------------------
// VQ: argmin over codebook (f32), z_q gather (f16 out), histogram, commitment.
// ---------------------------------------------------------------------------
__global__ __launch_bounds__(256)
void vq_kernel(const float* __restrict__ ze, const float* __restrict__ cb,
               _Float16* __restrict__ zq, int* __restrict__ idx_out,
               float* __restrict__ hist, float* __restrict__ accum)
{
    __shared__ float zrow[E_];
    __shared__ float bestd[256];
    __shared__ int   besti[256];
    int b = blockIdx.x, tid = threadIdx.x;
    for (int i = tid; i < E_; i += 256) zrow[i] = ze[(size_t)b * E_ + i];
    __syncthreads();
    float bd = 1e30f; int bi = 0;
    for (int k = tid; k < KCODE_; k += 256) {
        const float* c = cb + (size_t)k * E_;
        float d = 0.f;
#pragma unroll 8
        for (int e = 0; e < E_; ++e) { float t = zrow[e] - c[e]; d += t * t; }
        if (d < bd) { bd = d; bi = k; }
    }
    bestd[tid] = bd; besti[tid] = bi;
    __syncthreads();
    for (int s = 128; s > 0; s >>= 1) {
        if (tid < s) {
            if (bestd[tid + s] < bestd[tid] ||
                (bestd[tid + s] == bestd[tid] && besti[tid + s] < besti[tid])) {
                bestd[tid] = bestd[tid + s]; besti[tid] = besti[tid + s];
            }
        }
        __syncthreads();
    }
    int best = besti[0];
    if (tid == 0) { idx_out[b] = best; atomicAdd(&hist[best], 1.0f); }
    const float* c = cb + (size_t)best * E_;
    float part = 0.f;
    for (int e = tid; e < E_; e += 256) {
        float q = c[e];
        zq[(size_t)b * E_ + e] = (_Float16)q;
        float t = q - zrow[e];
        part += t * t;
    }
    __syncthreads();
    bestd[tid] = part;
    __syncthreads();
    for (int s = 128; s > 0; s >>= 1) {
        if (tid < s) bestd[tid] += bestd[tid + s];
        __syncthreads();
    }
    if (tid == 0) atomicAdd(accum, bestd[0]);
}

__global__ __launch_bounds__(256)
void fin_kernel(const float* __restrict__ hist, const float* __restrict__ accum,
                float* __restrict__ out_comm, float* __restrict__ out_perp)
{
    __shared__ float red[256];
    int tid = threadIdx.x;
    float s = 0.f;
    for (int k = tid; k < KCODE_; k += 256) {
        float p = hist[k] / (float)B_;
        s += -p * __logf(p + 1e-10f);
    }
    red[tid] = s; __syncthreads();
    for (int st = 128; st > 0; st >>= 1) {
        if (tid < st) red[tid] += red[tid + st];
        __syncthreads();
    }
    if (tid == 0) {
        *out_perp = __expf(red[0]);
        *out_comm = 0.1f * accum[0] / (float)(B_ * E_);
    }
}

// velocity head: out[b, 0..1] = d2[b, :128] @ Wd3[j, :128] + bd3[j]
__global__ __launch_bounds__(64)
void head_kernel(const _Float16* __restrict__ X, const float* __restrict__ W,
                 const float* __restrict__ bias, float* __restrict__ out)
{
    int b = blockIdx.x;
    int lane = threadIdx.x & 31;
    int j = threadIdx.x >> 5;
    float s = 0.f;
    for (int c = lane; c < 128; c += 32)
        s += (float)X[(size_t)b * 128 + c] * W[j * 128 + c];
    for (int off = 1; off < 32; off <<= 1) s += __shfl_xor(s, off, 32);
    if (lane == 0) out[b * 2 + j] = s + bias[j];
}

// ---------------------------------------------------------------------------
static inline void* ws_take(char*& p, size_t bytes)
{
    void* r = (void*)p;
    p += (bytes + 255) & ~(size_t)255;
    return r;
}

extern "C" void kernel_launch(void* const* d_in, const int* in_sizes, int n_in,
                              void* d_out, int out_size, void* d_ws, size_t ws_size,
                              hipStream_t stream)
{
    (void)in_sizes; (void)n_in; (void)out_size; (void)ws_size;

    const float* x     = (const float*)d_in[0];
    const float* W_in  = (const float*)d_in[1];
    const float* b_in  = (const float*)d_in[2];
    const float* cls   = (const float*)d_in[3];
    const float* Wqkv  = (const float*)d_in[4];
    const float* bqkv  = (const float*)d_in[5];
    const float* Wo    = (const float*)d_in[6];
    const float* bo    = (const float*)d_in[7];
    const float* W1    = (const float*)d_in[8];
    const float* b1    = (const float*)d_in[9];
    const float* W2    = (const float*)d_in[10];
    const float* b2    = (const float*)d_in[11];
    const float* ln1_g = (const float*)d_in[12];
    const float* ln1_b = (const float*)d_in[13];
    const float* ln2_g = (const float*)d_in[14];
    const float* ln2_b = (const float*)d_in[15];
    const float* lnf_g = (const float*)d_in[16];
    const float* lnf_b = (const float*)d_in[17];
    const float* W_out = (const float*)d_in[18];
    const float* b_out = (const float*)d_in[19];
    const float* cb    = (const float*)d_in[20];
    const float* Wd1   = (const float*)d_in[21];
    const float* bd1   = (const float*)d_in[22];
    const float* lnd_g = (const float*)d_in[23];
    const float* lnd_b = (const float*)d_in[24];
    const float* Wd2   = (const float*)d_in[25];
    const float* bd2   = (const float*)d_in[26];
    const float* Wd3   = (const float*)d_in[27];
    const float* bd3   = (const float*)d_in[28];

    // ---- workspace layout ----
    char* p = (char*)d_ws;
    float*    pe     = (float*)   ws_take(p, sizeof(float) * S_ * D_);
    _Float16* xh     = (_Float16*)ws_take(p, sizeof(_Float16) * (size_t)B_ * S_ * C_);
    _Float16* Winh   = (_Float16*)ws_take(p, sizeof(_Float16) * D_ * C_);
    _Float16* Wqkvh  = (_Float16*)ws_take(p, sizeof(_Float16) * (size_t)L_ * 3 * D_ * D_);
    _Float16* Woh    = (_Float16*)ws_take(p, sizeof(_Float16) * (size_t)L_ * D_ * D_);
    _Float16* W1h    = (_Float16*)ws_take(p, sizeof(_Float16) * (size_t)L_ * FF_ * D_);
    _Float16* W2h    = (_Float16*)ws_take(p, sizeof(_Float16) * (size_t)L_ * D_ * FF_);
    _Float16* Wouth  = (_Float16*)ws_take(p, sizeof(_Float16) * E_ * D_);
    _Float16* Wd1h   = (_Float16*)ws_take(p, sizeof(_Float16) * D_ * E_);
    _Float16* Wd2h   = (_Float16*)ws_take(p, sizeof(_Float16) * E_ * D_);
    _Float16* hbuf   = (_Float16*)ws_take(p, sizeof(_Float16) * (size_t)B_ * T_ * D_);
    _Float16* qkvbuf = (_Float16*)ws_take(p, sizeof(_Float16) * (size_t)B_ * T_ * 3 * D_);
    _Float16* obuf   = (_Float16*)ws_take(p, sizeof(_Float16) * (size_t)B_ * T_ * D_);
    _Float16* tmp    = (_Float16*)ws_take(p, sizeof(_Float16) * (size_t)B_ * T_ * D_);
    _Float16* ffbuf  = (_Float16*)ws_take(p, sizeof(_Float16) * (size_t)B_ * T_ * FF_);
    _Float16* clsln  = (_Float16*)ws_take(p, sizeof(_Float16) * (size_t)B_ * D_);
    _Float16* dbuf   = (_Float16*)ws_take(p, sizeof(_Float16) * (size_t)B_ * D_);
    _Float16* d2buf  = (_Float16*)ws_take(p, sizeof(_Float16) * (size_t)B_ * E_);
    _Float16* zqbuf  = (_Float16*)ws_take(p, sizeof(_Float16) * (size_t)B_ * E_);
    float*    zebuf  = (float*)   ws_take(p, sizeof(float) * (size_t)B_ * E_);
    float*    hist   = (float*)   ws_take(p, sizeof(float) * KCODE_);
    float*    accum  = (float*)   ws_take(p, sizeof(float) * 16);

    const int Mt = B_ * T_;
    auto cvt = [&](const float* s, _Float16* d, size_t n) {
        cvt_kernel<<<(unsigned)((n + 255) / 256), 256, 0, stream>>>(s, d, n);
    };

    // ---- pre-convert inputs/weights to f16 ----
    cvt(x,     xh,    (size_t)B_ * S_ * C_);
    cvt(W_in,  Winh,  (size_t)D_ * C_);
    cvt(Wqkv,  Wqkvh, (size_t)L_ * 3 * D_ * D_);
    cvt(Wo,    Woh,   (size_t)L_ * D_ * D_);
    cvt(W1,    W1h,   (size_t)L_ * FF_ * D_);
    cvt(W2,    W2h,   (size_t)L_ * D_ * FF_);
    cvt(W_out, Wouth, (size_t)E_ * D_);
    cvt(Wd1,   Wd1h,  (size_t)D_ * E_);
    cvt(Wd2,   Wd2h,  (size_t)E_ * D_);
    pe_kernel<<<(S_ * (D_ / 2) + 255) / 256, 256, 0, stream>>>(pe);

    // ---- input projection + PE, remapped to [B, T, D] rows 1..64 ----
    {
        int M = B_ * S_;
        dim3 g(M / 64, D_ / 64);
        gemm_wmma_kernel<<<g, 256, 0, stream>>>(xh, C_, Winh, C_, b_in, pe,
                                                hbuf, D_, M, D_, C_, 2, 1);
    }
    cls_kernel<<<(B_ * D_ + 255) / 256, 256, 0, stream>>>(cls, hbuf);

    for (int l = 0; l < L_; ++l) {
        const _Float16* Wq  = Wqkvh + (size_t)l * 3 * D_ * D_;
        const float*    bq  = bqkv  + (size_t)l * 3 * D_;
        const _Float16* Wol = Woh   + (size_t)l * D_ * D_;
        const float*    bol = bo    + (size_t)l * D_;
        const _Float16* W1l = W1h   + (size_t)l * FF_ * D_;
        const float*    b1l = b1    + (size_t)l * FF_;
        const _Float16* W2l = W2h   + (size_t)l * D_ * FF_;
        const float*    b2l = b2    + (size_t)l * D_;

        { dim3 g(Mt / 64, (3 * D_) / 64);
          gemm_wmma_kernel<<<g, 256, 0, stream>>>(hbuf, D_, Wq, D_, bq, nullptr,
                                                  qkvbuf, 3 * D_, Mt, 3 * D_, D_, 0, 1); }

        attn_kernel<<<B_ * H_, 160, 0, stream>>>(qkvbuf, obuf);

        { dim3 g(Mt / 64, D_ / 64);
          gemm_wmma_kernel<<<g, 256, 0, stream>>>(obuf, D_, Wol, D_, bol, hbuf,
                                                  tmp, D_, Mt, D_, D_, 3, 1); }
        ln_kernel<<<(Mt + 7) / 8, 256, 0, stream>>>(tmp, D_, ln1_g + l * D_,
                                                    ln1_b + l * D_, hbuf, D_, Mt, D_, 0);

        { dim3 g(Mt / 64, FF_ / 64);
          gemm_wmma_kernel<<<g, 256, 0, stream>>>(hbuf, D_, W1l, D_, b1l, nullptr,
                                                  ffbuf, FF_, Mt, FF_, D_, 1, 1); }
        { dim3 g(Mt / 64, D_ / 64);
          gemm_wmma_kernel<<<g, 256, 0, stream>>>(ffbuf, FF_, W2l, FF_, b2l, hbuf,
                                                  tmp, D_, Mt, D_, FF_, 3, 1); }
        ln_kernel<<<(Mt + 7) / 8, 256, 0, stream>>>(tmp, D_, ln2_g + l * D_,
                                                    ln2_b + l * D_, hbuf, D_, Mt, D_, 0);
    }

    // ---- final LN over cls rows -> z_e ----
    ln_kernel<<<(B_ + 7) / 8, 256, 0, stream>>>(hbuf, (size_t)T_ * D_, lnf_g, lnf_b,
                                                clsln, D_, B_, D_, 0);
    { dim3 g(B_ / 64, E_ / 64);
      gemm_wmma_kernel<<<g, 256, 0, stream>>>(clsln, D_, Wouth, D_, b_out, nullptr,
                                              zebuf, E_, B_, E_, D_, 0, 0); }

    // ---- VQ ----
    float* fout = (float*)d_out;
    int* idx_out = (int*)(fout + (size_t)B_ * 2);
    init_kernel<<<(KCODE_ + 255) / 256, 256, 0, stream>>>(hist, accum);
    vq_kernel<<<B_, 256, 0, stream>>>(zebuf, cb, zqbuf, idx_out, hist, accum);

    // ---- decoder ----
    { dim3 g(B_ / 64, D_ / 64);
      gemm_wmma_kernel<<<g, 256, 0, stream>>>(zqbuf, E_, Wd1h, E_, bd1, nullptr,
                                              clsln, D_, B_, D_, E_, 0, 1); }
    ln_kernel<<<(B_ + 7) / 8, 256, 0, stream>>>(clsln, D_, lnd_g, lnd_b,
                                                dbuf, D_, B_, D_, 1);
    { dim3 g(B_ / 64, E_ / 64);
      gemm_wmma_kernel<<<g, 256, 0, stream>>>(dbuf, D_, Wd2h, D_, bd2, nullptr,
                                              d2buf, E_, B_, E_, D_, 1, 1); }
    head_kernel<<<B_, 64, 0, stream>>>(d2buf, Wd3, bd3, fout);

    fin_kernel<<<1, 256, 0, stream>>>(hist, accum,
                                      fout + (size_t)B_ * 2 + B_,
                                      fout + (size_t)B_ * 2 + B_ + 1);
}
